// PndModelLSTM_8899172237339
// MI455X (gfx1250) — matrix operand order repaired and checked by reading, not compiled, not run on hardware
//
#include <hip/hip_runtime.h>
#include <hip/hip_bf16.h>
#include <cstdint>

// ---------------------------------------------------------------------------
// 3-layer LSTM (H=64) over S=2048 steps, B=512.
// - Recurrent GEMMs on v_wmma_f32_16x16x32_f16 (CDNA5 wave32 WMMA).
// - Layer 0: K=4 input projection folded into the WMMA K-dim by extending the
//   recurrent-state LDS rows to [h(64) | x_t(4) | 0(28)]  (K = 96, 6 WMMA/step).
// - Layers 1/2: x-activations prefetched one step ahead into double-buffered
//   LDS via GLOBAL_LOAD_ASYNC_TO_LDS_B128 (ASYNCcnt), K = 128, 8 WMMA/step.
// ---------------------------------------------------------------------------

typedef __attribute__((ext_vector_type(16))) _Float16 v16h;
typedef __attribute__((ext_vector_type(8)))  _Float16 v8h;
typedef __attribute__((ext_vector_type(4)))  _Float16 v4h;
typedef __attribute__((ext_vector_type(8)))  float    v8f;
typedef __attribute__((ext_vector_type(4)))  float    v4f;

#define SEQ   2048
#define BATCH 512
#define H     64
#define G4    256      // 4*H, gate width (i,f,g,o)
#define ROWS  16       // batch rows per workgroup (one WMMA M-tile)
#define NT    256      // threads per block = 8 wave32

__device__ __forceinline__ float sigmoidf_fast(float x) {
    return 1.0f / (1.0f + __expf(-x));
}
__device__ __forceinline__ float tanhf_fast(float x) {
    float e = __expf(-2.0f * x);
    return (1.0f - e) / (1.0f + e);
}

// 16-byte async global->LDS copy tracked by ASYNCcnt. The low 32 bits of a
// flat pointer into LDS are the LDS byte offset (ISA 10.2 aperture calc).
__device__ __forceinline__ void async_copy_b128(_Float16* lds_dst,
                                                const _Float16* gsrc) {
    unsigned lds_off = (unsigned)(uintptr_t)lds_dst;
    asm volatile("global_load_async_to_lds_b128 %0, %1, off"
                 :: "v"(lds_off), "v"(gsrc) : "memory");
}
__device__ __forceinline__ void wait_asynccnt0() {
#if __has_builtin(__builtin_amdgcn_s_wait_asynccnt)
    __builtin_amdgcn_s_wait_asynccnt(0);
#else
    asm volatile("s_wait_asynccnt 0" ::: "memory");
#endif
}

// A-fragment (16x32 f16, M x K), contiguous K order (matches B):
//   lanes 0-15  (row = lane)    : elems 0-15 -> K kbase+0..15
//   lanes 16-31 (row = lane-16) : elems 0-15 -> K kbase+16..31
// Two adjacent 16B chunks -> two ds_load_b128 into adjacent VGPR quads.
__device__ __forceinline__ v16h load_a_frag(const _Float16* rowp, int kbase, int khalf) {
    const v8h lo = *(const v8h*)(rowp + kbase + khalf * 16);
    const v8h hi = *(const v8h*)(rowp + kbase + 8 + khalf * 16);
    v16h r;
#pragma unroll
    for (int i = 0; i < 8; ++i) { r[i] = lo[i]; r[i + 8] = hi[i]; }
    return r;
}

// B-fragment (32x16 f16, K x N): lane = N column (0-15 / 16-31 split by K half);
// element i -> K = kbase + (lane>>4)*16 + i. Source weights are f32 row-major
// W[n][k] (gates = act @ W^T), converted to f16 once before the time loop.
__device__ __forceinline__ v16h load_b_frag(const float* __restrict__ W, int ldw,
                                            int n, int kbase, int khalf) {
    const float* p = W + (size_t)n * ldw + kbase + khalf * 16;
    v16h r;
#pragma unroll
    for (int i = 0; i < 16; ++i) r[i] = (_Float16)p[i];
    return r;
}

// Zero-padded B-fragment for layer 0's K=4 input weights (k >= kin -> 0).
__device__ __forceinline__ v16h load_b_frag_pad(const float* __restrict__ W, int kin,
                                                int n, int khalf) {
    v16h r;
#pragma unroll
    for (int i = 0; i < 16; ++i) {
        const int kl = khalf * 16 + i;
        r[i] = (kl < kin) ? (_Float16)W[n * kin + kl] : (_Float16)0.0f;
    }
    return r;
}

template <bool LAYER0>
__global__ __launch_bounds__(NT)
void lstm_layer_kernel(const float*    __restrict__ x0,    // layer0: (B, S, 4) f32
                       const _Float16* __restrict__ xact,  // layers1/2: (S, B, H) f16
                       const float*    __restrict__ Wih,   // (G4, Kin)
                       const float*    __restrict__ Whh,   // (G4, H)
                       const float*    __restrict__ bih,   // (G4,)
                       const float*    __restrict__ bhh,   // (G4,)
                       _Float16*       __restrict__ yact)  // (S, B, H) f16
{
    // layer0 rows carry [h(64) | x_t(4) | 0(28)]  -> K = 96
    constexpr int HR = LAYER0 ? 96 : H;

    __shared__ alignas(16) _Float16 lds_h[ROWS][HR];       // recurrent state (+x for L0)
    __shared__ alignas(16) _Float16 lds_x[2][ROWS][H];     // x-tile double buffer (L1/2)
    __shared__ alignas(16) float    lds_g[ROWS][G4];       // 16 KB gate pre-activations

    const int tid    = threadIdx.x;
    const int lane   = tid & 31;
    const int wave   = tid >> 5;            // 0..7: gate-column slice owner
    const int wgbase = blockIdx.x * ROWS;   // batch-row tile base

    const int nn    = lane & 15;            // N within tile
    const int khalf = lane >> 4;            // K half selector
    const int arow  = lane & 15;            // A-fragment row
    const int nbase = wave * 32;            // this wave's gate columns [nbase, nbase+32)

    // async x-tile prefetch assignment: threads 0..127 move 16B each (2 KB tile)
    const int  pf_row   = tid >> 3;
    const int  pf_chunk = (tid & 7) * 8;
    const bool pf_on    = tid < (ROWS * H * 2 / 16);   // 128 threads

    // ---- one-time: build weight B-fragments in registers --------------------
    v16h bh[2][2];                          // Whh [k-chunk][n-tile]
    v16h bx[2][2];                          // Wih full-K (layers 1/2)
    v16h b2[2];                             // Wih0 zero-padded K-chunk (layer 0)
#pragma unroll
    for (int nt = 0; nt < 2; ++nt) {
        const int n = nbase + nt * 16 + nn;
#pragma unroll
        for (int kc = 0; kc < 2; ++kc) {
            bh[kc][nt] = load_b_frag(Whh, H, n, kc * 32, khalf);
            if constexpr (!LAYER0) bx[kc][nt] = load_b_frag(Wih, H, n, kc * 32, khalf);
        }
        if constexpr (LAYER0) b2[nt] = load_b_frag_pad(Wih, 4, n, khalf);
    }
    const float bias0 = bih[nbase + nn]      + bhh[nbase + nn];
    const float bias1 = bih[nbase + 16 + nn] + bhh[nbase + 16 + nn];

    for (int i = tid; i < ROWS * HR; i += NT) ((_Float16*)lds_h)[i] = (_Float16)0.0f;

    // per-thread cell state: 4 consecutive hidden units of one batch row
    const int erow = tid >> 4;              // 0..15
    const int ehid = (tid & 15) * 4;        // 0,4,...,60
    float c0 = 0.f, c1 = 0.f, c2 = 0.f, c3 = 0.f;

    if constexpr (!LAYER0) {                // prefetch x-tile for t = 0
        if (pf_on) {
            async_copy_b128(&lds_x[0][pf_row][pf_chunk],
                            xact + ((size_t)0 * BATCH + wgbase + pf_row) * H + pf_chunk);
        }
        wait_asynccnt0();
    }
    __syncthreads();                        // lds_h zeros visible before x_0 stage

    if constexpr (LAYER0) {                 // stage x_0 into lds_h[.][64..67]
        if ((tid & 15) == 0) {
            const v4f xv = *(const v4f*)(x0 + ((size_t)(wgbase + erow)) * SEQ * 4);
            v4h xq;
#pragma unroll
            for (int k = 0; k < 4; ++k) xq[k] = (_Float16)xv[k];
            *(v4h*)&lds_h[erow][H] = xq;
        }
        __syncthreads();
    }

    for (int t = 0; t < SEQ; ++t) {
        const int buf = t & 1;

        // ---- A fragments, all from LDS --------------------------------------
        v16h ah0 = load_a_frag(&lds_h[arow][0], 0,  khalf);
        v16h ah1 = load_a_frag(&lds_h[arow][0], 32, khalf);
        v16h ax0, ax1;
        if constexpr (LAYER0) {
            ax0 = load_a_frag(&lds_h[arow][0], 64, khalf);  // [x_t | zeros]
        } else {
            ax0 = load_a_frag(&lds_x[buf][arow][0], 0,  khalf);
            ax1 = load_a_frag(&lds_x[buf][arow][0], 32, khalf);
        }

        v8f acc0, acc1;                      // bias-initialized accumulators
#pragma unroll
        for (int v = 0; v < 8; ++v) { acc0[v] = bias0; acc1[v] = bias1; }

        // gates[16 x 32] = x_t @ Wih^T + h @ Whh^T + b   (K = 96 / 128)
        acc0 = __builtin_amdgcn_wmma_f32_16x16x32_f16(false, ah0, false, bh[0][0], (short)0, acc0, false, false);
        acc0 = __builtin_amdgcn_wmma_f32_16x16x32_f16(false, ah1, false, bh[1][0], (short)0, acc0, false, false);
        acc1 = __builtin_amdgcn_wmma_f32_16x16x32_f16(false, ah0, false, bh[0][1], (short)0, acc1, false, false);
        acc1 = __builtin_amdgcn_wmma_f32_16x16x32_f16(false, ah1, false, bh[1][1], (short)0, acc1, false, false);
        if constexpr (LAYER0) {
            acc0 = __builtin_amdgcn_wmma_f32_16x16x32_f16(false, ax0, false, b2[0], (short)0, acc0, false, false);
            acc1 = __builtin_amdgcn_wmma_f32_16x16x32_f16(false, ax0, false, b2[1], (short)0, acc1, false, false);
        } else {
            acc0 = __builtin_amdgcn_wmma_f32_16x16x32_f16(false, ax0, false, bx[0][0], (short)0, acc0, false, false);
            acc0 = __builtin_amdgcn_wmma_f32_16x16x32_f16(false, ax1, false, bx[1][0], (short)0, acc0, false, false);
            acc1 = __builtin_amdgcn_wmma_f32_16x16x32_f16(false, ax0, false, bx[0][1], (short)0, acc1, false, false);
            acc1 = __builtin_amdgcn_wmma_f32_16x16x32_f16(false, ax1, false, bx[1][1], (short)0, acc1, false, false);
        }

        // ---- overlap: async-prefetch next timestep's x-tile (L1/2) ----------
        if constexpr (!LAYER0) {
            if (t + 1 < SEQ && pf_on) {
                async_copy_b128(&lds_x[buf ^ 1][pf_row][pf_chunk],
                                xact + ((size_t)(t + 1) * BATCH + wgbase + pf_row) * H
                                     + pf_chunk);
            }
        }

        // ---- scatter C-fragments to LDS (row = v + 8*khalf, col = nn) -------
#pragma unroll
        for (int v = 0; v < 8; ++v) {
            lds_g[v + khalf * 8][nbase + nn]      = acc0[v];
            lds_g[v + khalf * 8][nbase + 16 + nn] = acc1[v];
        }
        __syncthreads();

        // ---- elementwise LSTM cell update (4 hidden units per thread) -------
        const v4f gi = *(const v4f*)&lds_g[erow][ehid];
        const v4f gf = *(const v4f*)&lds_g[erow][H + ehid];
        const v4f gg = *(const v4f*)&lds_g[erow][2 * H + ehid];
        const v4f go = *(const v4f*)&lds_g[erow][3 * H + ehid];

        float cc[4] = {c0, c1, c2, c3};
        v4h hq;
#pragma unroll
        for (int j = 0; j < 4; ++j) {
            const float ig = sigmoidf_fast(gi[j]);
            const float fg = sigmoidf_fast(gf[j]);
            const float gt = tanhf_fast(gg[j]);
            const float og = sigmoidf_fast(go[j]);
            cc[j] = fg * cc[j] + ig * gt;
            hq[j] = (_Float16)(og * tanhf_fast(cc[j]));
        }
        c0 = cc[0]; c1 = cc[1]; c2 = cc[2]; c3 = cc[3];

        *(v4h*)&lds_h[erow][ehid] = hq;                                  // recurrent state
        *(v4h*)(yact + ((size_t)t * BATCH + wgbase + erow) * H + ehid) = hq;  // next layer

        if constexpr (LAYER0) {              // stage x_{t+1} into lds_h[.][64..67]
            if ((tid & 15) == 0 && t + 1 < SEQ) {
                const v4f xv = *(const v4f*)(x0 + ((size_t)(wgbase + erow)) * SEQ * 4
                                                + (size_t)(t + 1) * 4);
                v4h xq;
#pragma unroll
                for (int k = 0; k < 4; ++k) xq[k] = (_Float16)xv[k];
                *(v4h*)&lds_h[erow][H] = xq;
            }
        }
        if constexpr (!LAYER0) wait_asynccnt0();   // t+1 x-tile landed in LDS
        __syncthreads();
    }
}

__global__ __launch_bounds__(256)
void fc_kernel(const _Float16* __restrict__ act,   // (S,B,H): read t = S-1
               const float* __restrict__ fc_w,     // (2, H)
               const float* __restrict__ fc_b,     // (2,)
               float* __restrict__ out)            // (B, 2)
{
    const int b = blockIdx.x * blockDim.x + threadIdx.x;
    if (b >= BATCH) return;
    const _Float16* hp = act + ((size_t)(SEQ - 1) * BATCH + b) * H;
    float s0 = fc_b[0], s1 = fc_b[1];
#pragma unroll 8
    for (int k = 0; k < H; ++k) {
        const float h = (float)hp[k];
        s0 += h * fc_w[k];
        s1 += h * fc_w[H + k];
    }
    out[b * 2 + 0] = s0;
    out[b * 2 + 1] = s1;
}

extern "C" void kernel_launch(void* const* d_in, const int* in_sizes, int n_in,
                              void* d_out, int out_size, void* d_ws, size_t ws_size,
                              hipStream_t stream) {
    (void)in_sizes; (void)n_in; (void)out_size; (void)ws_size;
    const float* x    = (const float*)d_in[0];
    const float* Wih0 = (const float*)d_in[1];
    const float* Whh0 = (const float*)d_in[2];
    const float* bih0 = (const float*)d_in[3];
    const float* bhh0 = (const float*)d_in[4];
    const float* Wih1 = (const float*)d_in[5];
    const float* Whh1 = (const float*)d_in[6];
    const float* bih1 = (const float*)d_in[7];
    const float* bhh1 = (const float*)d_in[8];
    const float* Wih2 = (const float*)d_in[9];
    const float* Whh2 = (const float*)d_in[10];
    const float* bih2 = (const float*)d_in[11];
    const float* bhh2 = (const float*)d_in[12];
    const float* fcw  = (const float*)d_in[13];
    const float* fcb  = (const float*)d_in[14];

    // ping-pong f16 activation buffers, (S,B,H) each = 128 MB
    _Float16* bufA = (_Float16*)d_ws;
    _Float16* bufB = bufA + (size_t)SEQ * BATCH * H;

    dim3 grid(BATCH / ROWS);   // 32 workgroups
    dim3 block(NT);            // 8 wave32 each

    lstm_layer_kernel<true ><<<grid, block, 0, stream>>>(x,       nullptr, Wih0, Whh0, bih0, bhh0, bufA);
    lstm_layer_kernel<false><<<grid, block, 0, stream>>>(nullptr, bufA,    Wih1, Whh1, bih1, bhh1, bufB);
    lstm_layer_kernel<false><<<grid, block, 0, stream>>>(nullptr, bufB,    Wih2, Whh2, bih2, bhh2, bufA);
    fc_kernel<<<(BATCH + 255) / 256, 256, 0, stream>>>(bufA, fcw, fcb, (float*)d_out);
}
